// CausalSelfAttention_16733192585503
// MI455X (gfx1250) — matrix operand order repaired and checked by reading, not compiled, not run on hardware
//
#include <hip/hip_runtime.h>
#include <hip/hip_bf16.h>

typedef __attribute__((ext_vector_type(16))) _Float16 v16h;
typedef __attribute__((ext_vector_type(8)))  _Float16 v8h;
typedef __attribute__((ext_vector_type(8)))  float    v8f;

#define D_MODEL  1024
#define N_HEADS  16
#define HEAD_DIM 64
#define SEQ_T    2048
#define BATCH    2
#define M_ROWS   (BATCH * SEQ_T)   // 4096

// ---------------- WMMA fragment helpers (contiguous-k layout, same k order for A and B) ----------------

// A fragment: 16x32 f16, row-major source with leading dim `ld` (elements).
// lane (0-15): row m=lane, K=0..15 contiguous
// lane (16-31): row m=lane-16, K=16..31 contiguous
__device__ __forceinline__ v16h load_A_frag(const _Float16* A, int ld) {
  const int lane = threadIdx.x & 31;
  const int m  = lane & 15;
  const int kh = (lane >> 4) << 4;          // 0 or 16
  const _Float16* p = A + (size_t)m * ld + kh;
  return *(const v16h*)(p);
}

// B fragment: 32x16 f16 from an N-major ("transposed") buffer Bt[n][k], ld in elements.
// lane (0-15): col n=lane, K=0..15 contiguous; lane (16-31): col n=lane-16, K=16..31.
__device__ __forceinline__ v16h load_B_frag(const _Float16* Bt, int ld) {
  const int lane = threadIdx.x & 31;
  const int n  = lane & 15;
  const int kh = (lane >> 4) << 4;          // 0 or 16
  return *(const v16h*)(Bt + (size_t)n * ld + kh);
}

__device__ __forceinline__ v8f wmma_f16(v16h a, v16h b, v8f c) {
  return __builtin_amdgcn_wmma_f32_16x16x32_f16(false, a, false, b, (short)0, c,
                                                false, false);
}

__device__ __forceinline__ float redmax16(float v) {
  v = fmaxf(v, __shfl_xor(v, 1));
  v = fmaxf(v, __shfl_xor(v, 2));
  v = fmaxf(v, __shfl_xor(v, 4));
  v = fmaxf(v, __shfl_xor(v, 8));
  return v;
}
__device__ __forceinline__ float redsum16(float v) {
  v += __shfl_xor(v, 1);
  v += __shfl_xor(v, 2);
  v += __shfl_xor(v, 4);
  v += __shfl_xor(v, 8);
  return v;
}

// ---------------- precision / layout staging kernels ----------------

__global__ __launch_bounds__(256) void cvt_f32_to_f16(const float* __restrict__ in,
                                                      _Float16* __restrict__ out, int n) {
  int i = blockIdx.x * 256 + threadIdx.x;
  if (i < n) out[i] = (_Float16)in[i];
}

// in: [R][C] f32 row-major; out: [C][R] f16 (N-major for B fragments)
__global__ __launch_bounds__(256) void transpose_f32_to_f16(const float* __restrict__ in,
                                                            _Float16* __restrict__ out,
                                                            int R, int C) {
  int c = blockIdx.x * 16 + threadIdx.x;
  int r = blockIdx.y * 16 + threadIdx.y;
  if (r < R && c < C) out[(size_t)c * R + r] = (_Float16)in[(size_t)r * C + c];
}

// ---------------- QKV projection GEMM: [4096,1024] x [1024,3072] ----------------
// Wave tile 64x64 (4 M-tiles x 4 N-tiles): 8 fragment loads per 16 WMMAs.
// Writes Q,K as [B*H][T][64] f16 and V transposed as [B*H][64][T] f16 so the
// attention B-fragments are contiguous 32B loads.
__global__ __launch_bounds__(128) void gemm_qkv(const _Float16* __restrict__ Xh,
                                                const _Float16* __restrict__ WqkvT,
                                                _Float16* __restrict__ Qb,
                                                _Float16* __restrict__ Kb,
                                                _Float16* __restrict__ Vt) {
  const int wave = threadIdx.x >> 5;
  const int lane = threadIdx.x & 31;
  const int row0 = blockIdx.x * 256 + wave * 64;
  const int col0 = blockIdx.y * 64;

  v8f acc[4][4];
#pragma unroll
  for (int mt = 0; mt < 4; ++mt)
#pragma unroll
    for (int nt = 0; nt < 4; ++nt) acc[mt][nt] = (v8f)(0.0f);

  for (int ks = 0; ks < D_MODEL; ks += 32) {
    v16h bf[4];
#pragma unroll
    for (int nt = 0; nt < 4; ++nt)
      bf[nt] = load_B_frag(WqkvT + (size_t)(col0 + nt * 16) * D_MODEL + ks, D_MODEL);
#pragma unroll
    for (int mt = 0; mt < 4; ++mt) {
      v16h a = load_A_frag(Xh + (size_t)(row0 + mt * 16) * D_MODEL + ks, D_MODEL);
#pragma unroll
      for (int nt = 0; nt < 4; ++nt) acc[mt][nt] = wmma_f16(a, bf[nt], acc[mt][nt]);
    }
  }

  const int ncol = lane & 15;
  const int mh   = (lane >> 4) * 8;
#pragma unroll
  for (int mt = 0; mt < 4; ++mt) {
#pragma unroll
    for (int nt = 0; nt < 4; ++nt) {
#pragma unroll
      for (int r = 0; r < 8; ++r) {
        const int rm = row0 + mt * 16 + mh + r;
        const int c  = col0 + nt * 16 + ncol;
        const int bb = rm >> 11;           // / SEQ_T
        const int t  = rm & (SEQ_T - 1);
        const int which = c >> 10;         // 0:Q 1:K 2:V
        const int cc = c & (D_MODEL - 1);
        const int h  = cc >> 6;
        const int d  = cc & (HEAD_DIM - 1);
        const int bh = bb * N_HEADS + h;
        const _Float16 hv = (_Float16)acc[mt][nt][r];
        if (which == 0)      Qb[((size_t)bh * SEQ_T + t) * HEAD_DIM + d] = hv;
        else if (which == 1) Kb[((size_t)bh * SEQ_T + t) * HEAD_DIM + d] = hv;
        else                 Vt[((size_t)bh * HEAD_DIM + d) * SEQ_T + t] = hv;
      }
    }
  }
}

// ---------------- flash attention: 1 wave = 16 queries, online softmax ----------------
__global__ __launch_bounds__(128) void attention_fwd(const _Float16* __restrict__ Qb,
                                                     const _Float16* __restrict__ Kb,
                                                     const _Float16* __restrict__ Vt,
                                                     _Float16* __restrict__ Attn) {
  __shared__ __align__(64) _Float16 Pl[4][16][32];   // per-wave P tile staging

  const int wave = threadIdx.x >> 5;
  const int lane = threadIdx.x & 31;
  const int bh   = blockIdx.y;
  const int bb   = bh >> 4;
  const int h    = bh & (N_HEADS - 1);
  const int q0   = blockIdx.x * 64 + wave * 16;

  const _Float16* Qp = Qb + ((size_t)bh * SEQ_T + q0) * HEAD_DIM;
  const _Float16* Kp = Kb + (size_t)bh * SEQ_T * HEAD_DIM;
  const _Float16* Vp = Vt + (size_t)bh * HEAD_DIM * SEQ_T;

  const v16h qf0 = load_A_frag(Qp, HEAD_DIM);        // hd 0..31
  const v16h qf1 = load_A_frag(Qp + 32, HEAD_DIM);   // hd 32..63

  v8f o[4];
#pragma unroll
  for (int nt = 0; nt < 4; ++nt) o[nt] = (v8f)(0.0f);
  float mrow[8], lrow[8];
#pragma unroll
  for (int r = 0; r < 8; ++r) { mrow[r] = -3.0e38f; lrow[r] = 0.0f; }

  const int moff = (lane >> 4) * 8;    // row half offset
  const int ncol = lane & 15;
  const float scale = 0.125f;          // 1/sqrt(64)

  for (int kb = 0; kb < q0 + 16; kb += 32) {
    // ---- S = Q K^T for 32 keys (two 16-wide tiles) ----
    v8f s0 = (v8f)(0.0f), s1 = (v8f)(0.0f);
    {
      v16h k0 = load_B_frag(Kp + (size_t)kb * HEAD_DIM, HEAD_DIM);
      v16h k1 = load_B_frag(Kp + (size_t)(kb + 16) * HEAD_DIM, HEAD_DIM);
      s0 = wmma_f16(qf0, k0, s0);
      s1 = wmma_f16(qf0, k1, s1);
      v16h k2 = load_B_frag(Kp + (size_t)kb * HEAD_DIM + 32, HEAD_DIM);
      v16h k3 = load_B_frag(Kp + (size_t)(kb + 16) * HEAD_DIM + 32, HEAD_DIM);
      s0 = wmma_f16(qf1, k2, s0);
      s1 = wmma_f16(qf1, k3, s1);
    }

    const bool need_mask = (kb + 31 > q0);
    float alpha[8];
#pragma unroll
    for (int r = 0; r < 8; ++r) {
      float a = s0[r] * scale;
      float c = s1[r] * scale;
      if (need_mask) {
        const int lim = q0 + moff + r;         // query index for this row
        if (kb + ncol > lim)      a = -3.0e38f;
        if (kb + 16 + ncol > lim) c = -3.0e38f;
      }
      float mx = redmax16(fmaxf(a, c));
      const float mnew = fmaxf(mrow[r], mx);
      const float al = __expf(mrow[r] - mnew);
      const float p0 = __expf(a - mnew);
      const float p1 = __expf(c - mnew);
      lrow[r] = lrow[r] * al + redsum16(p0 + p1);
      mrow[r] = mnew;
      alpha[r] = al;
      const int m = moff + r;
      Pl[wave][m][ncol]      = (_Float16)p0;   // same-wave DS: in order
      Pl[wave][m][ncol + 16] = (_Float16)p1;
    }

    // ---- O = O*alpha + P @ V ----
#pragma unroll
    for (int nt = 0; nt < 4; ++nt)
#pragma unroll
      for (int r = 0; r < 8; ++r) o[nt][r] *= alpha[r];

    const v16h pf = load_A_frag(&Pl[wave][0][0], 32);
#pragma unroll
    for (int nt = 0; nt < 4; ++nt) {
      v16h vf = load_B_frag(Vp + (size_t)(nt * 16) * SEQ_T + kb, SEQ_T);
      o[nt] = wmma_f16(pf, vf, o[nt]);
    }
  }

  // ---- normalize and write [B][T][C] f16 (head-interleaved for out-proj) ----
#pragma unroll
  for (int r = 0; r < 8; ++r) {
    const float inv = 1.0f / lrow[r];
    const int q = q0 + moff + r;
#pragma unroll
    for (int nt = 0; nt < 4; ++nt) {
      Attn[((size_t)(bb * SEQ_T + q)) * D_MODEL + h * HEAD_DIM + nt * 16 + ncol] =
          (_Float16)(o[nt][r] * inv);
    }
  }
}

// ---------------- output projection: [4096,1024] x [1024,1024] -> f32 ----------------
__global__ __launch_bounds__(128) void gemm_out(const _Float16* __restrict__ Ah,
                                                const _Float16* __restrict__ WoT,
                                                float* __restrict__ Y) {
  const int wave = threadIdx.x >> 5;
  const int lane = threadIdx.x & 31;
  const int row0 = blockIdx.x * 256 + wave * 64;
  const int col0 = blockIdx.y * 64;

  v8f acc[4][4];
#pragma unroll
  for (int mt = 0; mt < 4; ++mt)
#pragma unroll
    for (int nt = 0; nt < 4; ++nt) acc[mt][nt] = (v8f)(0.0f);

  for (int ks = 0; ks < D_MODEL; ks += 32) {
    v16h bf[4];
#pragma unroll
    for (int nt = 0; nt < 4; ++nt)
      bf[nt] = load_B_frag(WoT + (size_t)(col0 + nt * 16) * D_MODEL + ks, D_MODEL);
#pragma unroll
    for (int mt = 0; mt < 4; ++mt) {
      v16h a = load_A_frag(Ah + (size_t)(row0 + mt * 16) * D_MODEL + ks, D_MODEL);
#pragma unroll
      for (int nt = 0; nt < 4; ++nt) acc[mt][nt] = wmma_f16(a, bf[nt], acc[mt][nt]);
    }
  }

  const int ncol = lane & 15;
  const int mh   = (lane >> 4) * 8;
#pragma unroll
  for (int mt = 0; mt < 4; ++mt)
#pragma unroll
    for (int nt = 0; nt < 4; ++nt)
#pragma unroll
      for (int r = 0; r < 8; ++r) {
        const int rm = row0 + mt * 16 + mh + r;
        const int c  = col0 + nt * 16 + ncol;
        Y[(size_t)rm * D_MODEL + c] = acc[mt][nt][r];
      }
}

// ---------------- host launch ----------------
extern "C" void kernel_launch(void* const* d_in, const int* in_sizes, int n_in,
                              void* d_out, int out_size, void* d_ws, size_t ws_size,
                              hipStream_t stream) {
  (void)in_sizes; (void)n_in; (void)out_size; (void)ws_size;
  const float* x     = (const float*)d_in[0];   // [2,2048,1024]
  const float* w_qkv = (const float*)d_in[1];   // [1024,3072]
  const float* w_out = (const float*)d_in[2];   // [1024,1024]
  float* y = (float*)d_out;                     // [2,2048,1024]

  char* ws = (char*)d_ws;
  size_t off = 0;
  auto alloc = [&](size_t bytes) -> void* {
    void* p = ws + off;
    off += (bytes + 255) & ~size_t(255);
    return p;
  };
  const size_t eBT = (size_t)M_ROWS * D_MODEL;           // 4M elements
  _Float16* Xh    = (_Float16*)alloc(eBT * 2);
  _Float16* WqkvT = (_Float16*)alloc((size_t)3 * D_MODEL * D_MODEL * 2);
  _Float16* WoT   = (_Float16*)alloc((size_t)D_MODEL * D_MODEL * 2);
  _Float16* Qb    = (_Float16*)alloc(eBT * 2);
  _Float16* Kb    = (_Float16*)alloc(eBT * 2);
  _Float16* Vt    = (_Float16*)alloc(eBT * 2);
  _Float16* Attn  = (_Float16*)alloc(eBT * 2);

  // stage precision/layouts
  cvt_f32_to_f16<<<(int)((eBT + 255) / 256), 256, 0, stream>>>(x, Xh, (int)eBT);
  transpose_f32_to_f16<<<dim3(3 * D_MODEL / 16, D_MODEL / 16), dim3(16, 16), 0, stream>>>(
      w_qkv, WqkvT, D_MODEL, 3 * D_MODEL);
  transpose_f32_to_f16<<<dim3(D_MODEL / 16, D_MODEL / 16), dim3(16, 16), 0, stream>>>(
      w_out, WoT, D_MODEL, D_MODEL);

  // qkv projection: grid (4096/256, 3072/64)
  gemm_qkv<<<dim3(M_ROWS / 256, 3 * D_MODEL / 64), 128, 0, stream>>>(Xh, WqkvT, Qb, Kb, Vt);

  // flash attention
  attention_fwd<<<dim3(SEQ_T / 64, BATCH * N_HEADS), 128, 0, stream>>>(Qb, Kb, Vt, Attn);

  // output projection: grid (4096/256, 1024/64)
  gemm_out<<<dim3(M_ROWS / 256, D_MODEL / 64), 128, 0, stream>>>(Attn, WoT, y);
}
